// InterFreqAttention_20452634263747
// MI455X (gfx1250) — matrix operand order repaired and checked by reading, not compiled, not run on hardware
//
#include <hip/hip_runtime.h>

#define DEV __device__ __forceinline__

typedef __attribute__((ext_vector_type(16))) __bf16 v16bf;
typedef __attribute__((ext_vector_type(8)))  float  v8f;

namespace {

constexpr int Cc = 64, Nn = 8, Ss = 1024, Dd = 256;

union BF16x16 { v16bf v; uint4 q[2]; unsigned short h[16]; };

// Native conversion: backend selects v_cvt_*bf16_f32 (RNE) instead of a
// 3-op integer round sequence.
DEV __bf16 f2bf(float f) { return (__bf16)f; }
DEV unsigned short f2bfu(float f) {
  __bf16 b = (__bf16)f;
  return __builtin_bit_cast(unsigned short, b);
}

#define WMMA_BF16(A, B, C) \
  __builtin_amdgcn_wmma_f32_16x16x32_bf16(false, (A), false, (B), (short)0, (C), false, false)

DEV float redsum16(float x) {
  #pragma unroll
  for (int m = 1; m < 16; m <<= 1) x += __shfl_xor(x, m, 32);
  return x;
}
DEV float redmax16(float x) {
  #pragma unroll
  for (int m = 1; m < 16; m <<= 1) x = fmaxf(x, __shfl_xor(x, m, 32));
  return x;
}

// B-tile (32x16, bf16) fetch from pre-packed weights: one contiguous 32B per lane.
DEV v16bf load_btile(const __bf16* __restrict__ packed, int Kt, int nt, int kt, int lane) {
  BF16x16 u;
  const uint4* p = (const uint4*)(packed + ((size_t)(nt * Kt + kt) * 32 + lane) * 16);
  u.q[0] = p[0]; u.q[1] = p[1];
  return u.v;
}

// B-tile from a row-major matrix where lane n = row (key/d), halves k contiguous.
DEV v16bf load_brow(const __bf16* row, int off) {
  BF16x16 u;
  const uint4* p = (const uint4*)(row + off);
  u.q[0] = p[0]; u.q[1] = p[1];
  return u.v;
}

// A-tile (16x32, bf16) from a row-major row pointer (global or LDS).
// halves 0..7 -> k = hi*8+0..7 ; halves 8..15 -> k = 16+hi*8+0..7
DEV v16bf load_atile(const __bf16* row, int dbase, int hi) {
  BF16x16 u;
  const uint4* p0 = (const uint4*)(row + dbase + hi * 8);
  const uint4* p1 = (const uint4*)(row + dbase + 16 + hi * 8);
  u.q[0] = *p0; u.q[1] = *p1;
  return u.v;
}

DEV void zero8(v8f& a) {
  #pragma unroll
  for (int e = 0; e < 8; ++e) a[e] = 0.0f;
}

} // namespace

// ---------------- weight packing into WMMA B-tile layout ----------------
__global__ void pack_b_kernel(const float* __restrict__ src, __bf16* __restrict__ dst,
                              int K, int N) {
  int idx = blockIdx.x * blockDim.x + threadIdx.x;
  if (idx >= K * N) return;
  int k = idx / N, n = idx % N;
  int kt = k >> 5, kin = k & 31, nt = n >> 4;
  int lane = (n & 15) | (((kin >> 3) & 1) << 4);
  int h = (kin & 7) | ((kin >> 4) << 3);
  int Kt = K >> 5;
  dst[(((size_t)(nt * Kt + kt) * 32 + lane) << 4) + h] = f2bf(src[idx]);
}

// ---------------- stage 1: Q/K/V projection + LayerNorm ----------------
__global__ __launch_bounds__(256)
void proj_kernel(const float* __restrict__ fea_lf, const float* __restrict__ fea_hf,
                 const __bf16* __restrict__ WpIn, const __bf16* __restrict__ WpKv,
                 const float* __restrict__ gamma, const float* __restrict__ beta,
                 __bf16* __restrict__ Qn, __bf16* __restrict__ Kn,
                 __bf16* __restrict__ Vt, float* __restrict__ Vrm) {
  const int wave = threadIdx.x >> 5;
  const int lane = threadIdx.x & 31;
  const int ln16 = lane & 15, hi = lane >> 4;
  const int bn = blockIdx.y, b = bn >> 3, nn = bn & 7;
  const int s0 = (blockIdx.x * 8 + wave) * 16;
  const int s = s0 + ln16;

  // A-tiles: 16 tokens x 32 channels, two chunks over C=64.
  v16bf a_lf[2], a_hf[2];
  #pragma unroll
  for (int kk = 0; kk < 2; ++kk) {
    BF16x16 ul, uh;
    #pragma unroll
    for (int h = 0; h < 16; ++h) {
      int c = kk * 32 + hi * 8 + (h & 7) + ((h & 8) ? 16 : 0);
      size_t idx = ((size_t)(b * Cc + c) * Nn + nn) * Ss + s;
      ul.h[h] = f2bfu(fea_lf[idx]);
      uh.h[h] = f2bfu(fea_hf[idx]);
    }
    a_lf[kk] = ul.v; a_hf[kk] = uh.v;
  }

  v8f acc[16];
  float sum[8], sq[8], mean[8], rstd[8];

  // ---- Q = LN(tok_lf @ W_in) ----
  #pragma unroll
  for (int t = 0; t < 16; ++t) {
    zero8(acc[t]);
    #pragma unroll
    for (int kk = 0; kk < 2; ++kk)
      acc[t] = WMMA_BF16(a_lf[kk], load_btile(WpIn, 2, t, kk, lane), acc[t]);
  }
  #pragma unroll
  for (int r = 0; r < 8; ++r) { sum[r] = 0.f; sq[r] = 0.f; }
  #pragma unroll
  for (int t = 0; t < 16; ++t)
    #pragma unroll
    for (int r = 0; r < 8; ++r) { float x = acc[t][r]; sum[r] += x; sq[r] += x * x; }
  #pragma unroll
  for (int r = 0; r < 8; ++r) {
    float su = redsum16(sum[r]), sdq = redsum16(sq[r]);
    mean[r] = su * (1.0f / Dd);
    float var = sdq * (1.0f / Dd) - mean[r] * mean[r];
    rstd[r] = rsqrtf(var + 1e-5f);
  }
  #pragma unroll
  for (int t = 0; t < 16; ++t) {
    float g = gamma[t * 16 + ln16], be = beta[t * 16 + ln16];
    #pragma unroll
    for (int r = 0; r < 8; ++r) {
      float y = (acc[t][r] - mean[r]) * rstd[r] * g + be;
      Qn[((size_t)bn * Ss + (s0 + r + 8 * hi)) * Dd + t * 16 + ln16] = f2bf(y);
    }
  }

  // ---- K = LN(tok_hf @ W_kv[:, :256]) ----
  #pragma unroll
  for (int t = 0; t < 16; ++t) {
    zero8(acc[t]);
    #pragma unroll
    for (int kk = 0; kk < 2; ++kk)
      acc[t] = WMMA_BF16(a_hf[kk], load_btile(WpKv, 2, t, kk, lane), acc[t]);
  }
  #pragma unroll
  for (int r = 0; r < 8; ++r) { sum[r] = 0.f; sq[r] = 0.f; }
  #pragma unroll
  for (int t = 0; t < 16; ++t)
    #pragma unroll
    for (int r = 0; r < 8; ++r) { float x = acc[t][r]; sum[r] += x; sq[r] += x * x; }
  #pragma unroll
  for (int r = 0; r < 8; ++r) {
    float su = redsum16(sum[r]), sdq = redsum16(sq[r]);
    mean[r] = su * (1.0f / Dd);
    float var = sdq * (1.0f / Dd) - mean[r] * mean[r];
    rstd[r] = rsqrtf(var + 1e-5f);
  }
  #pragma unroll
  for (int t = 0; t < 16; ++t) {
    float g = gamma[t * 16 + ln16], be = beta[t * 16 + ln16];
    #pragma unroll
    for (int r = 0; r < 8; ++r) {
      float y = (acc[t][r] - mean[r]) * rstd[r] * g + be;
      Kn[((size_t)bn * Ss + (s0 + r + 8 * hi)) * Dd + t * 16 + ln16] = f2bf(y);
    }
  }

  // ---- V = tok_hf @ W_kv[:, 256:] (no LN); store row-major f32 + transposed bf16 ----
  #pragma unroll
  for (int t = 0; t < 16; ++t) {
    zero8(acc[t]);
    #pragma unroll
    for (int kk = 0; kk < 2; ++kk)
      acc[t] = WMMA_BF16(a_hf[kk], load_btile(WpKv, 2, 16 + t, kk, lane), acc[t]);
    #pragma unroll
    for (int r = 0; r < 8; ++r) {
      int srow = s0 + r + 8 * hi;
      int d = t * 16 + ln16;
      Vrm[((size_t)bn * Ss + srow) * Dd + d] = acc[t][r];
      Vt[((size_t)bn * Dd + d) * Ss + srow] = f2bf(acc[t][r]);
    }
  }
}

// ---------------- stage 2: causal flash attention ----------------
__global__ __launch_bounds__(256)
void attn_kernel(const __bf16* __restrict__ Qn, const __bf16* __restrict__ Kn,
                 const __bf16* __restrict__ Vt, float* __restrict__ Att) {
  __shared__ alignas(16) __bf16 pbuf[8][16 * 32];
  const int wave = threadIdx.x >> 5;
  const int lane = threadIdx.x & 31;
  const int ln16 = lane & 15, hi = lane >> 4;
  const int bn = blockIdx.y;
  const int q0 = (blockIdx.x * 8 + wave) * 16;

  const __bf16* Qrow = Qn + ((size_t)bn * Ss + (q0 + ln16)) * Dd;
  v16bf qa[8];
  #pragma unroll
  for (int kk = 0; kk < 8; ++kk) qa[kk] = load_atile(Qrow, kk * 32, hi);

  v8f acc[16];
  #pragma unroll
  for (int t = 0; t < 16; ++t) zero8(acc[t]);
  float mrow[8], lrow[8];
  #pragma unroll
  for (int r = 0; r < 8; ++r) { mrow[r] = -1e30f; lrow[r] = 0.0f; }

  const int nblk = (q0 + 16 + 31) >> 5;  // causal: keys 0 .. q0+15
  for (int jb = 0; jb < nblk; ++jb) {
    const int kbase = jb * 32;
    v8f st0, st1; zero8(st0); zero8(st1);
    const __bf16* Krow0 = Kn + ((size_t)bn * Ss + (kbase + ln16)) * Dd;
    const __bf16* Krow1 = Krow0 + (size_t)16 * Dd;

    // Prefetch next K block while this one feeds the WMMAs (global_prefetch_b8).
    if (jb + 1 < nblk) {
      const __bf16* pf = Krow0 + (size_t)32 * Dd;
      __builtin_prefetch(pf, 0, 1);
      __builtin_prefetch(pf + (size_t)16 * Dd, 0, 1);
    }

    #pragma unroll
    for (int kk = 0; kk < 8; ++kk) {
      st0 = WMMA_BF16(qa[kk], load_atile(Krow0, kk * 32, hi), st0);
      st1 = WMMA_BF16(qa[kk], load_atile(Krow1, kk * 32, hi), st1);
    }
    float sc[8];
    #pragma unroll
    for (int r = 0; r < 8; ++r) {
      int qi = q0 + r + 8 * hi;
      int k0 = kbase + ln16, k1 = k0 + 16;
      float s0v = (k0 <= qi) ? st0[r] : -1e15f;
      float s1v = (k1 <= qi) ? st1[r] : -1e15f;
      float mx = redmax16(fmaxf(s0v, s1v));
      float nmr = fmaxf(mrow[r], mx);
      float scale = __expf(mrow[r] - nmr);
      float p0 = __expf(s0v - nmr), p1 = __expf(s1v - nmr);
      lrow[r] = lrow[r] * scale + redsum16(p0 + p1);
      mrow[r] = nmr; sc[r] = scale;
      pbuf[wave][(r + 8 * hi) * 32 + ln16]      = f2bf(p0);
      pbuf[wave][(r + 8 * hi) * 32 + 16 + ln16] = f2bf(p1);
    }
    #pragma unroll
    for (int t = 0; t < 16; ++t)
      #pragma unroll
      for (int r = 0; r < 8; ++r) acc[t][r] *= sc[r];
    asm volatile("s_wait_dscnt 0" ::: "memory");
    v16bf pa = load_atile(&pbuf[wave][ln16 * 32], 0, hi);
    #pragma unroll
    for (int t = 0; t < 16; ++t) {
      const __bf16* Vrow = Vt + ((size_t)bn * Dd + (t * 16 + ln16)) * Ss;
      acc[t] = WMMA_BF16(pa, load_atile(Vrow, kbase, hi), acc[t]);
    }
  }

  float rl[8];
  #pragma unroll
  for (int r = 0; r < 8; ++r) rl[r] = 1.0f / lrow[r];
  #pragma unroll
  for (int t = 0; t < 16; ++t)
    #pragma unroll
    for (int r = 0; r < 8; ++r)
      Att[((size_t)bn * Ss + q0 + r + 8 * hi) * Dd + t * 16 + ln16] = acc[t][r] * rl[r];
}

// ---------------- stage 3: LN2 + FFN + residual + out projection ----------------
__global__ __launch_bounds__(64)
void ffn_kernel(const float* __restrict__ Att, const float* __restrict__ Vrm,
                const __bf16* __restrict__ Wp1, const __bf16* __restrict__ Wp2,
                const __bf16* __restrict__ WpOut,
                const float* __restrict__ g2, const float* __restrict__ b2,
                float* __restrict__ out) {
  __shared__ alignas(16) __bf16 h1buf[2][16 * 512];
  const int wave = threadIdx.x >> 5;
  const int lane = threadIdx.x & 31;
  const int ln16 = lane & 15, hi = lane >> 4;
  const int bn = blockIdx.y, b = bn >> 3, nn = bn & 7;
  const int s0 = (blockIdx.x * 2 + wave) * 16;

  // LN2 stats for row (s0 + ln16): each lane sums half the row via b128 loads,
  // pair-reduce via xor 16.
  const float* arow = Att + ((size_t)bn * Ss + s0 + ln16) * Dd;
  const float4* arow4 = (const float4*)(arow + hi * 128);
  float sum = 0.f, sq = 0.f;
  #pragma unroll
  for (int i = 0; i < 32; ++i) {
    float4 x = arow4[i];
    sum += x.x + x.y + x.z + x.w;
    sq  += x.x * x.x + x.y * x.y + x.z * x.z + x.w * x.w;
  }
  sum += __shfl_xor(sum, 16, 32);
  sq  += __shfl_xor(sq, 16, 32);
  float mean = sum * (1.0f / Dd);
  float rstd = rsqrtf(sq * (1.0f / Dd) - mean * mean + 1e-5f);

  // normalized x as A-tiles (8 chunks of 32 over D), gathered with float4 loads
  v16bf xa[8];
  #pragma unroll
  for (int kk = 0; kk < 8; ++kk) {
    const float4* p0 = (const float4*)(arow + kk * 32 + hi * 8);
    const float4* p1 = (const float4*)(arow + kk * 32 + 16 + hi * 8);
    float v0[8], d0[8];
    float4 a0 = p0[0], a1 = p0[1], c0 = p1[0], c1 = p1[1];
    v0[0] = a0.x; v0[1] = a0.y; v0[2] = a0.z; v0[3] = a0.w;
    v0[4] = a1.x; v0[5] = a1.y; v0[6] = a1.z; v0[7] = a1.w;
    d0[0] = c0.x; d0[1] = c0.y; d0[2] = c0.z; d0[3] = c0.w;
    d0[4] = c1.x; d0[5] = c1.y; d0[6] = c1.z; d0[7] = c1.w;
    BF16x16 u;
    #pragma unroll
    for (int e = 0; e < 8; ++e) {
      int da = kk * 32 + hi * 8 + e;
      int db = kk * 32 + 16 + hi * 8 + e;
      u.h[e]     = f2bfu((v0[e] - mean) * rstd * g2[da] + b2[da]);
      u.h[8 + e] = f2bfu((d0[e] - mean) * rstd * g2[db] + b2[db]);
    }
    xa[kk] = u.v;
  }

  // h1 = relu(x @ W1)  -> LDS (row-major 16x512 bf16)
  for (int nt = 0; nt < 32; ++nt) {
    v8f a; zero8(a);
    #pragma unroll
    for (int kk = 0; kk < 8; ++kk)
      a = WMMA_BF16(xa[kk], load_btile(Wp1, 8, nt, kk, lane), a);
    #pragma unroll
    for (int r = 0; r < 8; ++r)
      h1buf[wave][(r + 8 * hi) * 512 + nt * 16 + ln16] = f2bf(fmaxf(a[r], 0.0f));
  }
  asm volatile("s_wait_dscnt 0" ::: "memory");

  // y = h1 @ W2 + V (residual)
  v8f acc[16];
  #pragma unroll
  for (int t = 0; t < 16; ++t) zero8(acc[t]);
  for (int kk = 0; kk < 16; ++kk) {
    v16bf ha = load_atile(&h1buf[wave][ln16 * 512], kk * 32, hi);
    #pragma unroll
    for (int dt = 0; dt < 16; ++dt)
      acc[dt] = WMMA_BF16(ha, load_btile(Wp2, 16, dt, kk, lane), acc[dt]);
  }
  #pragma unroll
  for (int dt = 0; dt < 16; ++dt)
    #pragma unroll
    for (int r = 0; r < 8; ++r)
      acc[dt][r] += Vrm[((size_t)bn * Ss + s0 + r + 8 * hi) * Dd + dt * 16 + ln16];

  // epi -> LDS (row-major 16x256 bf16, reuse h1buf)
  asm volatile("s_wait_dscnt 0" ::: "memory");
  #pragma unroll
  for (int dt = 0; dt < 16; ++dt)
    #pragma unroll
    for (int r = 0; r < 8; ++r)
      h1buf[wave][(r + 8 * hi) * 256 + dt * 16 + ln16] = f2bf(acc[dt][r]);
  asm volatile("s_wait_dscnt 0" ::: "memory");

  // out = epi @ W_out, scatter to (b, c, n, v, w)
  v8f oc[4];
  #pragma unroll
  for (int ct = 0; ct < 4; ++ct) zero8(oc[ct]);
  for (int kk = 0; kk < 8; ++kk) {
    v16bf ea = load_atile(&h1buf[wave][ln16 * 256], kk * 32, hi);
    #pragma unroll
    for (int ct = 0; ct < 4; ++ct)
      oc[ct] = WMMA_BF16(ea, load_btile(WpOut, 8, ct, kk, lane), oc[ct]);
  }
  #pragma unroll
  for (int ct = 0; ct < 4; ++ct)
    #pragma unroll
    for (int r = 0; r < 8; ++r) {
      int c = ct * 16 + ln16;
      int srow = s0 + r + 8 * hi;
      out[((size_t)(b * Cc + c) * Nn + nn) * Ss + srow] = oc[ct][r];
    }
}

// ---------------- launcher ----------------
extern "C" void kernel_launch(void* const* d_in, const int* in_sizes, int n_in,
                              void* d_out, int out_size, void* d_ws, size_t ws_size,
                              hipStream_t stream) {
  const float* fea_lf = (const float*)d_in[0];
  const float* fea_hf = (const float*)d_in[1];
  const float* W_in   = (const float*)d_in[2];
  const float* W_kv   = (const float*)d_in[3];
  const float* gamma  = (const float*)d_in[4];
  const float* beta   = (const float*)d_in[5];
  const float* g2     = (const float*)d_in[6];
  const float* b2     = (const float*)d_in[7];
  const float* W1     = (const float*)d_in[8];
  const float* W2     = (const float*)d_in[9];
  const float* W_out  = (const float*)d_in[10];
  float* out = (float*)d_out;

  char* ws = (char*)d_ws;
  size_t off = 0;
  auto alloc = [&](size_t bytes) -> char* {
    char* p = ws + off;
    off += (bytes + 255) & ~(size_t)255;
    return p;
  };
  __bf16* WpIn  = (__bf16*)alloc((size_t)64 * 256 * 2);
  __bf16* WpKv  = (__bf16*)alloc((size_t)64 * 512 * 2);
  __bf16* Wp1   = (__bf16*)alloc((size_t)256 * 512 * 2);
  __bf16* Wp2   = (__bf16*)alloc((size_t)512 * 256 * 2);
  __bf16* WpOut = (__bf16*)alloc((size_t)256 * 64 * 2);
  __bf16* Qn    = (__bf16*)alloc((size_t)32 * 1024 * 256 * 2);
  __bf16* Kn    = (__bf16*)alloc((size_t)32 * 1024 * 256 * 2);
  __bf16* Vt    = (__bf16*)alloc((size_t)32 * 256 * 1024 * 2);
  float*  Vrm   = (float*) alloc((size_t)32 * 1024 * 256 * 4);
  float*  Atb   = (float*) alloc((size_t)32 * 1024 * 256 * 4);
  (void)ws_size; (void)in_sizes; (void)n_in; (void)out_size;

  auto pack = [&](const float* src, __bf16* dst, int K, int N) {
    int total = K * N;
    pack_b_kernel<<<(total + 255) / 256, 256, 0, stream>>>(src, dst, K, N);
  };
  pack(W_in,  WpIn,  64, 256);
  pack(W_kv,  WpKv,  64, 512);
  pack(W1,    Wp1,   256, 512);
  pack(W2,    Wp2,   512, 256);
  pack(W_out, WpOut, 256, 64);

  proj_kernel<<<dim3(8, 32), 256, 0, stream>>>(fea_lf, fea_hf, WpIn, WpKv,
                                               gamma, beta, Qn, Kn, Vt, Vrm);
  attn_kernel<<<dim3(8, 32), 256, 0, stream>>>(Qn, Kn, Vt, Atb);
  ffn_kernel<<<dim3(32, 32), 64, 0, stream>>>(Atb, Vrm, Wp1, Wp2, WpOut, g2, b2, out);
}
